// AtomTransformer_5059471475248
// MI455X (gfx1250) — compile-verified, matrix operand order checked by reading, not run on hardware
//
#include <hip/hip_runtime.h>
#include <hip/hip_bf16.h>

// ---------------------------------------------------------------------------
// AtomTransformer forward, CDNA5 (gfx1250) wave32 + WMMA f16->f32.
//   B=4, N=2048, HID=256, H=4 heads x HEAD=64, LAYERS=4, FFN=512.
// Strategy:
//   - A[h,b,i,j] = exp(-d2/ls2[h]) materialized once in f16 (128 MiB),
//     row-normalization folded into per-row fp32 scale 1/(sum+1e-8) in the
//     attention epilogue.
//   - All big GEMMs via v_wmma_f32_16x16x32_f16, 256x64 block tile,
//     8 waves/block, each wave owns two 16-row strips (8 accumulators) so
//     every B-fragment is reused twice; B-tiles double-buffered in LDS.
//   - B-tile staging uses GLOBAL_LOAD_ASYNC_TO_LDS_B128 (ASYNCcnt) when the
//     toolchain exposes the builtin, else global->reg->ds_store fallback.
// ---------------------------------------------------------------------------

typedef __attribute__((ext_vector_type(16))) _Float16 v16h;
typedef __attribute__((ext_vector_type(8)))  _Float16 h8;
typedef __attribute__((ext_vector_type(8)))  float    v8f;
typedef int v4i __attribute__((__vector_size__(4 * sizeof(int))));

#define BB      4
#define NN      2048
#define BN      (BB * NN)
#define HID     256
#define HH      4
#define HEAD    64
#define NLAYER  4
#define FFN     512
#define IN_DIM  70

#if defined(__has_builtin)
#if __has_builtin(__builtin_amdgcn_global_load_async_to_lds_b128) && \
    __has_builtin(__builtin_amdgcn_s_wait_asynccnt)
#define USE_ASYNC_LDS 1
#endif
#endif
#ifndef USE_ASYNC_LDS
#define USE_ASYNC_LDS 0
#endif

// Build a 16x32 f16 A/B WMMA fragment from two contiguous 8-half chunks.
// Per ISA layout: lane l (row/col = l&15, g = l>>4), elements 0..7 <-> K =
// k0 + g*8 + 0..7 ; elements 8..15 <-> K = k0 + 16 + g*8 + 0..7.
__device__ __forceinline__ v16h mk_frag(const _Float16* __restrict__ p) {
  h8 lo = *(const h8*)p;
  h8 hi = *(const h8*)(p + 16);
  v16h f;
#pragma unroll
  for (int i = 0; i < 8; ++i) { f[i] = lo[i]; f[i + 8] = hi[i]; }
  return f;
}

// Stage one b128 chunk of the 64x32 B-tile into LDS (one chunk per thread).
__device__ __forceinline__ void stage_chunk(const _Float16* __restrict__ src,
                                            _Float16* dst) {
#if USE_ASYNC_LDS
  __builtin_amdgcn_global_load_async_to_lds_b128(
      (__attribute__((address_space(1))) v4i*)src,
      (__attribute__((address_space(3))) v4i*)dst, 0, 0);
#else
  *(h8*)dst = *(const h8*)src;
#endif
}

__device__ __forceinline__ void stage_sync() {
#if USE_ASYNC_LDS
  __builtin_amdgcn_s_wait_asynccnt(0);
#endif
  __syncthreads();
}

// Compute a 256(M) x 64(N) tile: 8 waves; wave w owns rows rowBase+16w and
// rowBase+128+16w (two 16-row strips, 8 accumulators). A row-major [*, lda];
// Bt row-major [Nout, ldb] (B transposed: k contiguous). K multiple of 32.
// bsh: double-buffered 64x32 f16 tiles.
__device__ __forceinline__ void tile256x64(
    const _Float16* __restrict__ A, int lda,
    const _Float16* __restrict__ Bt, int ldb,
    int K, int rowBase, int nb0,
    _Float16 (*bsh)[64 * 32], v8f acc[2][4]) {
  const int tid  = threadIdx.x;
  const int lane = tid & 31;
  const int wave = tid >> 5;
  const int g    = lane >> 4;
  const int l16  = lane & 15;
  const int colL = tid >> 2;   // 0..63: B-tile column this thread stages
  const int segL = tid & 3;    // 0..3 : 8-half segment within 32-k row
  const size_t bOff = (size_t)(nb0 + colL) * ldb + segL * 8;
  const int lof = colL * 32 + segL * 8;

  const v8f vzero = {};
#pragma unroll
  for (int s = 0; s < 2; ++s)
#pragma unroll
    for (int t = 0; t < 4; ++t) acc[s][t] = vzero;

  stage_chunk(&Bt[bOff], &bsh[0][lof]);
  stage_sync();

  const int KT = K >> 5;
  for (int kt = 0; kt < KT; ++kt) {
    const int cur = kt & 1;
    if (kt + 1 < KT)  // prefetch next B-tile into the other buffer
      stage_chunk(&Bt[bOff + (size_t)(kt + 1) * 32], &bsh[cur ^ 1][lof]);

    const int k0 = kt * 32;
    v16h af0 = mk_frag(&A[(size_t)(rowBase + wave * 16 + l16) * lda + k0 + g * 8]);
    v16h af1 = mk_frag(&A[(size_t)(rowBase + 128 + wave * 16 + l16) * lda + k0 + g * 8]);
#pragma unroll
    for (int t = 0; t < 4; ++t) {
      v16h bf = mk_frag(&bsh[cur][(t * 16 + l16) * 32 + g * 8]);
      acc[0][t] = __builtin_amdgcn_wmma_f32_16x16x32_f16(
          false, af0, false, bf, (short)0, acc[0][t], false, false);
      acc[1][t] = __builtin_amdgcn_wmma_f32_16x16x32_f16(
          false, af1, false, bf, (short)0, acc[1][t], false, false);
    }
    stage_sync();
  }
}

enum { MODE_RELU_F16 = 0, MODE_ADDRES_F32 = 1, MODE_VPROJ = 2 };

// Generic GEMM: C = A[M,K] x Bt[Nout,K]^T + bias, epilogue per mode.
// grid = (Nout/64, M/256), block = 256.
__global__ void __launch_bounds__(256)
wmma_gemm_kernel(const _Float16* __restrict__ A,
                 const _Float16* __restrict__ Bt,
                 const float* __restrict__ bias,
                 const float* __restrict__ res,
                 float* __restrict__ outF,
                 _Float16* __restrict__ outH,
                 int K, int Nout, int mode) {
  __shared__ __align__(16) _Float16 bsh[2][64 * 32];
  const int nb0     = blockIdx.x * 64;
  const int rowBase = blockIdx.y * 256;
  v8f acc[2][4];
  tile256x64(A, K, Bt, K, K, rowBase, nb0, bsh, acc);

  const int lane = threadIdx.x & 31;
  const int wave = threadIdx.x >> 5;
  const int g = lane >> 4, l16 = lane & 15;
#pragma unroll
  for (int s = 0; s < 2; ++s) {
    const int m0 = rowBase + s * 128 + wave * 16;
#pragma unroll
    for (int t = 0; t < 4; ++t) {
      const int n = nb0 + t * 16 + l16;
      const float bn = bias ? bias[n] : 0.f;
#pragma unroll
      for (int r = 0; r < 8; ++r) {
        const int m = m0 + g * 8 + r;          // C layout: VGPR r -> M = r+8g
        float v = acc[s][t][r] + bn;
        if (mode == MODE_RELU_F16) {
          v = fmaxf(v, 0.f);
          outH[(size_t)m * Nout + n] = (_Float16)v;
        } else if (mode == MODE_ADDRES_F32) {
          outF[(size_t)m * Nout + n] = v + res[(size_t)m * Nout + n];
        } else {  // MODE_VPROJ: scatter to Vt[h][b][d][j]
          const int b = m >> 11, j = m & (NN - 1);
          const int h = n >> 6,  d = n & 63;
          outH[((((size_t)h * BB + b) * HEAD + d) << 11) + j] = (_Float16)v;
        }
      }
    }
  }
}

// Attention: for z = h*B+b, C[i,d] = sum_j expA[z][i,j] * Vt[z][d,j],
// scaled by rscale[z][i]. grid = (N/256, H*B), block = 256.
__global__ void __launch_bounds__(256)
attn_wmma_kernel(const _Float16* __restrict__ Aall,
                 const _Float16* __restrict__ Vt,
                 const float* __restrict__ rscale,
                 _Float16* __restrict__ attOut) {
  __shared__ __align__(16) _Float16 bsh[2][64 * 32];
  const int z = blockIdx.y;
  const int h = z >> 2, b = z & 3;
  const _Float16* A  = Aall + (size_t)z * NN * NN;
  const _Float16* Bt = Vt + (size_t)z * HEAD * NN;
  const int rowBase = blockIdx.x * 256;
  v8f acc[2][4];
  tile256x64(A, NN, Bt, NN, NN, rowBase, 0, bsh, acc);

  const int lane = threadIdx.x & 31;
  const int wave = threadIdx.x >> 5;
  const int g = lane >> 4, l16 = lane & 15;
#pragma unroll
  for (int s = 0; s < 2; ++s) {
    const int m0 = rowBase + s * 128 + wave * 16;
#pragma unroll
    for (int t = 0; t < 4; ++t) {
      const int d = t * 16 + l16;
#pragma unroll
      for (int r = 0; r < 8; ++r) {
        const int i = m0 + g * 8 + r;
        const float rs = rscale[(size_t)z * NN + i];
        attOut[((size_t)(b * NN + i)) * HID + h * HEAD + d] =
            (_Float16)(acc[s][t][r] * rs);
      }
    }
  }
}

// A[h,b,i,j] = exp(-d2(b,i,j)/ls2[h]) in f16. 2^26 threads.
__global__ void a_fill_kernel(const float* __restrict__ xc,
                              _Float16* __restrict__ Aout) {
  const size_t idx = (size_t)blockIdx.x * blockDim.x + threadIdx.x;
  const int j = (int)(idx & (NN - 1));
  const int i = (int)((idx >> 11) & (NN - 1));
  const int b = (int)((idx >> 22) & 3);
  const int h = (int)(idx >> 24);
  const float inv_ls2[4] = {4.f, 1.f, 0.25f, 0.0625f};  // 1/ls^2
  const float* xi = xc + ((size_t)b * NN + i) * 3;
  const float* xj = xc + ((size_t)b * NN + j) * 3;
  const float dx = xi[0] - xj[0], dy = xi[1] - xj[1], dz = xi[2] - xj[2];
  const float d2 = dx * dx + dy * dy + dz * dz;
  Aout[idx] = (_Float16)__expf(-d2 * inv_ls2[h]);
}

// rscale[row] = 1/(sum_j A[row][j] + 1e-8), row in [0, H*B*N)
__global__ void __launch_bounds__(256)
rowsum_kernel(const _Float16* __restrict__ Aall, float* __restrict__ rscale) {
  const _Float16* a = Aall + (size_t)blockIdx.x * NN;
  const int j0 = threadIdx.x * 8;            // 256 threads * 8 halves = 2048
  h8 v = *(const h8*)&a[j0];
  float s = 0.f;
#pragma unroll
  for (int i = 0; i < 8; ++i) s += (float)v[i];
  for (int off = 16; off; off >>= 1) s += __shfl_xor(s, off, 32);
  __shared__ float sh[8];
  if ((threadIdx.x & 31) == 0) sh[threadIdx.x >> 5] = s;
  __syncthreads();
  if (threadIdx.x == 0) {
    float t = 0.f;
#pragma unroll
    for (int w = 0; w < 8; ++w) t += sh[w];
    rscale[blockIdx.x] = 1.f / (t + 1e-8f);
  }
}

// feats = relu(concat(xc, emb, lat) @ in_W + in_b). One block per row.
__global__ void __launch_bounds__(256)
input_mlp_kernel(const float* __restrict__ xc, const float* __restrict__ emb,
                 const float* __restrict__ lat, const float* __restrict__ W,
                 const float* __restrict__ bias, float* __restrict__ fF,
                 _Float16* __restrict__ fH) {
  __shared__ float xin[IN_DIM];
  const int m = blockIdx.x, n = threadIdx.x;
  if (n < 3)        xin[n] = xc[(size_t)m * 3 + n];
  else if (n < 67)  xin[n] = emb[(size_t)m * 64 + (n - 3)];
  else if (n < 70)  xin[n] = lat[(size_t)m * 3 + (n - 67)];
  __syncthreads();
  float acc = bias[n];
  for (int k = 0; k < IN_DIM; ++k) acc += xin[k] * W[k * HID + n];
  acc = fmaxf(acc, 0.f);
  fF[(size_t)m * HID + n] = acc;
  fH[(size_t)m * HID + n] = (_Float16)acc;
}

// LayerNorm over HID=256. One block per row, writes f32 + f16 copies.
__global__ void __launch_bounds__(256)
ln_kernel(const float* __restrict__ x, const float* __restrict__ gg,
          const float* __restrict__ bb, float* __restrict__ fF,
          _Float16* __restrict__ fH) {
  const int m = blockIdx.x, n = threadIdx.x;
  const float v = x[(size_t)m * HID + n];
  float s = v, q = v * v;
  for (int off = 16; off; off >>= 1) {
    s += __shfl_xor(s, off, 32);
    q += __shfl_xor(q, off, 32);
  }
  __shared__ float shs[8], shq[8];
  if ((n & 31) == 0) { shs[n >> 5] = s; shq[n >> 5] = q; }
  __syncthreads();
  s = 0.f; q = 0.f;
#pragma unroll
  for (int w = 0; w < 8; ++w) { s += shs[w]; q += shq[w]; }
  const float mu  = s * (1.f / HID);
  const float var = q * (1.f / HID) - mu * mu;
  const float y = (v - mu) * rsqrtf(var + 1e-5f) * gg[n] + bb[n];
  fF[(size_t)m * HID + n] = y;
  fH[(size_t)m * HID + n] = (_Float16)y;
}

// out[m, 0..2] = h1[m,:] @ out_W2 + out_b2 (fp32)
__global__ void out2_kernel(const _Float16* __restrict__ h1,
                            const float* __restrict__ W2,
                            const float* __restrict__ b2,
                            float* __restrict__ out) {
  const int m = blockIdx.x * blockDim.x + threadIdx.x;
  float a0 = b2[0], a1 = b2[1], a2 = b2[2];
  const _Float16* row = h1 + (size_t)m * HID;
  for (int k = 0; k < HID; ++k) {
    const float v = (float)row[k];
    a0 += v * W2[k * 3 + 0];
    a1 += v * W2[k * 3 + 1];
    a2 += v * W2[k * 3 + 2];
  }
  out[(size_t)m * 3 + 0] = a0;
  out[(size_t)m * 3 + 1] = a1;
  out[(size_t)m * 3 + 2] = a2;
}

// dst[mat][n*K+k] = f16(src[mat][k*Nout+n])  (weight transpose+convert)
__global__ void transpose_w_kernel(const float* __restrict__ src,
                                   _Float16* __restrict__ dst, int K, int Nout) {
  const size_t idx = (size_t)blockIdx.x * blockDim.x + threadIdx.x;
  const size_t per = (size_t)K * Nout;
  const size_t mat = idx / per;
  const size_t rem = idx - mat * per;
  const int n = (int)(rem / K);
  const int k = (int)(rem - (size_t)n * K);
  dst[mat * per + (size_t)n * K + k] =
      (_Float16)src[mat * per + (size_t)k * Nout + n];
}

// val_W [L][H][HID][HEAD] -> valWt [L][h*64+d][HID]  (concat heads, transposed)
__global__ void valw_prep_kernel(const float* __restrict__ src,
                                 _Float16* __restrict__ dst) {
  const size_t idx = (size_t)blockIdx.x * blockDim.x + threadIdx.x;  // 2^18
  const int d  = (int)(idx & 63);
  const int k  = (int)((idx >> 6) & 255);
  const int hh = (int)((idx >> 14) & 3);
  const int l  = (int)(idx >> 16);
  dst[((size_t)l * 256 + hh * 64 + d) * 256 + k] = (_Float16)src[idx];
}

extern "C" void kernel_launch(void* const* d_in, const int* in_sizes, int n_in,
                              void* d_out, int out_size, void* d_ws,
                              size_t ws_size, hipStream_t stream) {
  (void)in_sizes; (void)n_in; (void)out_size; (void)ws_size;
  const float* latent = (const float*)d_in[0];
  const float* xc     = (const float*)d_in[1];
  const float* emb    = (const float*)d_in[2];
  const float* in_W   = (const float*)d_in[3];
  const float* in_b   = (const float*)d_in[4];
  const float* val_W  = (const float*)d_in[5];
  const float* val_b  = (const float*)d_in[6];
  const float* attn_W = (const float*)d_in[7];
  const float* attn_b = (const float*)d_in[8];
  const float* ln1_g  = (const float*)d_in[9];
  const float* ln1_b  = (const float*)d_in[10];
  const float* ln2_g  = (const float*)d_in[11];
  const float* ln2_b  = (const float*)d_in[12];
  const float* ffn_W1 = (const float*)d_in[13];
  const float* ffn_b1 = (const float*)d_in[14];
  const float* ffn_W2 = (const float*)d_in[15];
  const float* ffn_b2 = (const float*)d_in[16];
  const float* out_W1 = (const float*)d_in[17];
  const float* out_b1 = (const float*)d_in[18];
  const float* out_W2 = (const float*)d_in[19];
  const float* out_b2 = (const float*)d_in[20];
  float* out = (float*)d_out;

  // ---- workspace carve-up (256B aligned) ----
  char* ws = (char*)d_ws;
  size_t off = 0;
  auto take = [&](size_t bytes) -> void* {
    void* p = (void*)(ws + off);
    off += (bytes + 255) & ~(size_t)255;
    return p;
  };
  _Float16* A16    = (_Float16*)take((size_t)HH * BB * NN * NN * 2);  // 128 MiB
  float*    rsc    = (float*)take((size_t)HH * BB * NN * 4);
  float*    featsF = (float*)take((size_t)BN * HID * 4);
  _Float16* featsH = (_Float16*)take((size_t)BN * HID * 2);
  _Float16* Vt     = (_Float16*)take((size_t)HH * BB * HEAD * NN * 2);
  _Float16* attH   = (_Float16*)take((size_t)BN * HID * 2);
  float*    tmpF   = (float*)take((size_t)BN * HID * 4);
  _Float16* hidH   = (_Float16*)take((size_t)BN * FFN * 2);
  _Float16* h1H    = (_Float16*)take((size_t)BN * HID * 2);
  _Float16* valWt  = (_Float16*)take((size_t)NLAYER * HID * HID * 2);
  _Float16* attnWt = (_Float16*)take((size_t)NLAYER * HID * HID * 2);
  _Float16* f1Wt   = (_Float16*)take((size_t)NLAYER * HID * FFN * 2);
  _Float16* f2Wt   = (_Float16*)take((size_t)NLAYER * FFN * HID * 2);
  _Float16* o1Wt   = (_Float16*)take((size_t)HID * HID * 2);

  // ---- weight prep (f32 -> f16 transposed) ----
  valw_prep_kernel<<<1024, 256, 0, stream>>>(val_W, valWt);
  transpose_w_kernel<<<1024, 256, 0, stream>>>(attn_W, attnWt, HID, HID);
  transpose_w_kernel<<<2048, 256, 0, stream>>>(ffn_W1, f1Wt, HID, FFN);
  transpose_w_kernel<<<2048, 256, 0, stream>>>(ffn_W2, f2Wt, FFN, HID);
  transpose_w_kernel<<<256, 256, 0, stream>>>(out_W1, o1Wt, HID, HID);

  // ---- input MLP + RBF attention weights ----
  input_mlp_kernel<<<BN, 256, 0, stream>>>(xc, emb, latent, in_W, in_b,
                                           featsF, featsH);
  a_fill_kernel<<<(1u << 26) / 256, 256, 0, stream>>>(xc, A16);
  rowsum_kernel<<<HH * BB * NN, 256, 0, stream>>>(A16, rsc);

  const dim3 gN256(HID / 64, BN / 256);   // Nout=256 GEMMs
  const dim3 gN512(FFN / 64, BN / 256);   // Nout=512 GEMM
  const dim3 gAttn(NN / 256, HH * BB);

  for (int l = 0; l < NLAYER; ++l) {
    // V = feats @ val_W[l] + val_b[l]  -> Vt[h][b][d][j] (f16)
    wmma_gemm_kernel<<<gN256, 256, 0, stream>>>(
        featsH, valWt + (size_t)l * HID * HID, val_b + l * HID,
        nullptr, nullptr, Vt, HID, HID, MODE_VPROJ);
    // att = diag(rsc) * expA @ V -> attH [B*N][H*HEAD] (f16)
    attn_wmma_kernel<<<gAttn, 256, 0, stream>>>(A16, Vt, rsc, attH);
    // tmp = feats + att @ attn_W[l] + attn_b[l]  (f32)
    wmma_gemm_kernel<<<gN256, 256, 0, stream>>>(
        attH, attnWt + (size_t)l * HID * HID, attn_b + l * HID,
        featsF, tmpF, nullptr, HID, HID, MODE_ADDRES_F32);
    // feats = LN1(tmp)
    ln_kernel<<<BN, 256, 0, stream>>>(tmpF, ln1_g + l * HID, ln1_b + l * HID,
                                      featsF, featsH);
    // hid = relu(feats @ ffn_W1[l] + b1)  (f16)
    wmma_gemm_kernel<<<gN512, 256, 0, stream>>>(
        featsH, f1Wt + (size_t)l * HID * FFN, ffn_b1 + l * FFN,
        nullptr, nullptr, hidH, HID, FFN, MODE_RELU_F16);
    // tmp = feats + hid @ ffn_W2[l] + b2  (f32)
    wmma_gemm_kernel<<<gN256, 256, 0, stream>>>(
        hidH, f2Wt + (size_t)l * FFN * HID, ffn_b2 + l * HID,
        featsF, tmpF, nullptr, FFN, HID, MODE_ADDRES_F32);
    // feats = LN2(tmp)
    ln_kernel<<<BN, 256, 0, stream>>>(tmpF, ln2_g + l * HID, ln2_b + l * HID,
                                      featsF, featsH);
  }

  // h1 = relu(feats @ out_W1 + out_b1); out = h1 @ out_W2 + out_b2
  wmma_gemm_kernel<<<gN256, 256, 0, stream>>>(
      featsH, o1Wt, out_b1, nullptr, nullptr, h1H, HID, HID, MODE_RELU_F16);
  out2_kernel<<<BN / 256, 256, 0, stream>>>(h1H, out_W2, out_b2, out);
}
